// EpisodicMemory_81381040324826
// MI455X (gfx1250) — matrix operand order repaired and checked by reading, not compiled, not run on hardware
//
#include <hip/hip_runtime.h>
#include <hip/hip_bf16.h>
#include <math.h>

#define BS_ 8
#define B_  8
#define M_  512
#define D_  128
#define N_  1024
#define NEGV (-1e30f)

typedef __attribute__((ext_vector_type(8)))  float  v8f;
typedef __attribute__((ext_vector_type(16))) __bf16 v16bf;
typedef __attribute__((ext_vector_type(8)))  __bf16 v8bf;

#if __has_builtin(__builtin_amdgcn_global_load_async_to_lds_b128)
#define HAVE_ASYNC_LDS 1
typedef int v4i_ __attribute__((vector_size(4 * sizeof(int))));
typedef __attribute__((address_space(1))) v4i_ as1_v4i;
typedef __attribute__((address_space(3))) v4i_ as3_v4i;
#endif

__device__ __forceinline__ int lane_id() { return (int)(threadIdx.x & 31u); }

__device__ __forceinline__ v8f wmma_bf16(v16bf a, v16bf b, v8f c) {
  return __builtin_amdgcn_wmma_f32_16x16x32_bf16(false, a, false, b, (short)0, c,
                                                 false, false);
}

__device__ __forceinline__ v16bf combine16(v8bf a, v8bf b) {
  return __builtin_shufflevector(a, b, 0, 1, 2, 3, 4, 5, 6, 7, 8, 9, 10, 11, 12,
                                 13, 14, 15);
}
// A-fragment (16x32 bf16): lane<16 row lr k={k0..k0+7,k0+16..k0+23};
// lane>=16 row lr k={k0+8..k0+15,k0+24..k0+31}. Works for LDS or global bf16.
__device__ __forceinline__ v16bf afrag_bf(const __bf16* base, int ld, int k0) {
  const int lr = lane_id() & 15, hf = lane_id() >> 4;
  const __bf16* p = base + (size_t)lr * ld + k0 + hf * 8;
  v8bf a = *(const v8bf*)p;
  v8bf b = *(const v8bf*)(p + 16);
  return combine16(a, b);
}
// B-fragment (32x16 bf16): lane holds column lr; lane<16 k={k0..k0+7,k0+16..k0+23},
// lane>=16 k={k0+8..k0+15,k0+24..k0+31}. Source stores each column's k-run contiguously.
__device__ __forceinline__ v16bf bfrag_bf(const __bf16* base, int ld, int k0) {
  const int lr = lane_id() & 15, hf = lane_id() >> 4;
  const __bf16* p = base + (size_t)lr * ld + k0 + hf * 8;
  v8bf a = *(const v8bf*)p;
  v8bf b = *(const v8bf*)(p + 16);
  return combine16(a, b);
}

// 16-byte global -> LDS copy; async on CDNA5 (ASYNCcnt), fallback ds_store.
__device__ __forceinline__ void cp16_g2l(__bf16* lds_dst, const __bf16* gsrc) {
#ifdef HAVE_ASYNC_LDS
  __builtin_amdgcn_global_load_async_to_lds_b128((as1_v4i*)gsrc,
                                                 (as3_v4i*)lds_dst, 0, 0);
#else
  *(v8bf*)lds_dst = *(const v8bf*)gsrc;
#endif
}
__device__ __forceinline__ void stage_fence() {
#ifdef HAVE_ASYNC_LDS
#if __has_builtin(__builtin_amdgcn_s_wait_asynccnt)
  __builtin_amdgcn_s_wait_asynccnt(0);
#else
  asm volatile("s_wait_asynccnt 0x0" ::: "memory");
#endif
#endif
  __syncthreads();
}

__device__ __forceinline__ float half_max(float v) {
#pragma unroll
  for (int m = 1; m < 16; m <<= 1) v = fmaxf(v, __shfl_xor(v, m, 32));
  return v;
}
__device__ __forceinline__ float half_sum(float v) {
#pragma unroll
  for (int m = 1; m < 16; m <<= 1) v += __shfl_xor(v, m, 32);
  return v;
}
__device__ __forceinline__ float softplus01(float x) { return log1pf(__expf(x)) + 0.1f; }

// -------------------- prep: w_norm + surprise magnitude --------------------
__global__ void k_prep(const float* __restrict__ w_cand,
                       const float* __restrict__ surprise,
                       float* __restrict__ wn, float* __restrict__ smag) {
  __shared__ float rw[4], rs[4];
  const int row = blockIdx.x;
  const int t = (int)threadIdx.x;  // 128 == D_
  float wv = w_cand[(size_t)row * D_ + t];
  float sv = surprise[(size_t)row * D_ + t];
  float a = wv * wv, c = sv * sv;
#pragma unroll
  for (int m = 1; m < 32; m <<= 1) { a += __shfl_xor(a, m, 32); c += __shfl_xor(c, m, 32); }
  if ((t & 31) == 0) { rw[t >> 5] = a; rs[t >> 5] = c; }
  __syncthreads();
  float wsum = rw[0] + rw[1] + rw[2] + rw[3];
  float ssum = rs[0] + rs[1] + rs[2] + rs[3];
  wn[(size_t)row * D_ + t] = wv / fmaxf(sqrtf(wsum), 1e-12f);
  if (t == 0) smag[row] = sqrtf(ssum);
}

__global__ void k_hasact(const float* __restrict__ emS, float* __restrict__ ha) {
  __shared__ int fl;
  if (threadIdx.x == 0) fl = 0;
  __syncthreads();
  const float* p = emS + (size_t)blockIdx.x * M_;
  int any = 0;
  for (int i = (int)threadIdx.x; i < M_; i += 256) any |= (p[i] > 0.f) ? 1 : 0;
  if (any) atomicOr(&fl, 1);
  __syncthreads();
  if (threadIdx.x == 0) ha[blockIdx.x] = fl ? 1.f : 0.f;
}

// -------------------- bf16 conversion / transposition into workspace ------
__global__ void k_cvt_copy(const float* __restrict__ src, __bf16* __restrict__ dst,
                           int count) {
  int i = (int)(blockIdx.x * 256 + threadIdx.x);
  if (i < count) dst[i] = (__bf16)src[i];
}
// per batch (blockIdx.y): dst[b][c][r] = src[b][r][c]
__global__ void k_cvt_tr(const float* __restrict__ src, __bf16* __restrict__ dst,
                         int rows, int cols) {
  size_t per = (size_t)rows * cols;
  size_t i = (size_t)blockIdx.x * 256 + threadIdx.x;
  if (i < per) {
    int r = (int)(i % rows);
    int c = (int)(i / rows);
    size_t bo = (size_t)blockIdx.y * per;
    dst[bo + (size_t)c * rows + r] = (__bf16)src[bo + (size_t)r * cols + c];
  }
}

// -------------------- fused 2-step attention (flash, bf16 WMMA) -----------
__global__ __launch_bounds__(256, 1)
void k_attn(const float* __restrict__ seed, const __bf16* __restrict__ emKb,
            const __bf16* __restrict__ emVt, const float* __restrict__ emS,
            const float* __restrict__ w1, const float* __restrict__ w2,
            const float* __restrict__ gb, const float* __restrict__ raw_tau,
            const float* __restrict__ hact_ws, float* __restrict__ y_em) {
  __shared__ __align__(16) __bf16 sh[24576];  // Kc|Vt|Pw (48KB); yW overlays Kc+Vt
  __shared__ float shS[64];
  const int blk = blockIdx.x;
  const int nt = blk & 7, b = (blk >> 3) & 7, bs = blk >> 6;
  const int wave = (int)(threadIdx.x >> 5);
  const int lr = lane_id() & 15, hf = lane_id() >> 4;
  const int row0 = nt * 128 + wave * 16;
  const float invtau = 1.0f / softplus01(raw_tau[b]);
  const float hact = hact_ws[bs * B_ + b];
  const float* seedB = seed + (size_t)bs * N_ * D_;
  const __bf16* Kg = emKb + (size_t)(bs * B_ + b) * M_ * D_;
  const __bf16* Vg = emVt + (size_t)(bs * B_ + b) * D_ * M_;  // [d][m]
  const float* Sb = emS + (size_t)(bs * B_ + b) * M_;

  float w1v[8], w2v[8], gbv[8];
#pragma unroll
  for (int f = 0; f < 8; ++f) {
    int d = f * 16 + lr;
    w1v[f] = w1[b * D_ + d]; w2v[f] = w2[b * D_ + d]; gbv[f] = gb[b * D_ + d];
  }

  float yc[8][8];  // C-layout: yc[f][r] = y[row0+r+8*hf][f*16+lr]
#pragma unroll
  for (int f = 0; f < 8; ++f)
#pragma unroll
    for (int r = 0; r < 8; ++r)
      yc[f][r] = seedB[(size_t)(row0 + r + 8 * hf) * D_ + f * 16 + lr];

  __bf16* Kc = sh;                        // [64][128]
  __bf16* Vt = sh + 8192;                 // [128][64]
  __bf16* Pw = sh + 16384 + wave * 1024;  // 16x64 per wave
  __bf16* yW = sh + wave * 2048;          // 16x128 per wave (overlay)

  for (int step = 0; step < 2; ++step) {
    __syncthreads();
#pragma unroll
    for (int f = 0; f < 8; ++f)
#pragma unroll
      for (int r = 0; r < 8; ++r)
        yW[(r + 8 * hf) * 128 + f * 16 + lr] = (__bf16)yc[f][r];
    __syncthreads();
    v16bf ya[4];
#pragma unroll
    for (int ka = 0; ka < 4; ++ka) ya[ka] = afrag_bf(yW, 128, ka * 32);

    float rowmax[8], rowsum[8];
    v8f acc[8];
#pragma unroll
    for (int r = 0; r < 8; ++r) { rowmax[r] = -INFINITY; rowsum[r] = 0.f; }
#pragma unroll
    for (int f = 0; f < 8; ++f)
#pragma unroll
      for (int r = 0; r < 8; ++r) acc[f][r] = 0.f;

    for (int mc = 0; mc < M_; mc += 64) {
      __syncthreads();  // previous chunk fully consumed
      for (int u = (int)threadIdx.x; u < 1024; u += 256) {
        // Kc: contiguous 8192 bf16 (64 rows x 128)
        cp16_g2l(Kc + u * 8, Kg + (size_t)mc * D_ + u * 8);
        // Vt: 128 rows of 64 (row = d)
        int d = u >> 3, q = u & 7;
        cp16_g2l(Vt + d * 64 + q * 8, Vg + (size_t)d * M_ + mc + q * 8);
      }
      if (threadIdx.x < 64) shS[threadIdx.x] = Sb[mc + threadIdx.x];
      stage_fence();

      v8f sc[4];
#pragma unroll
      for (int ct = 0; ct < 4; ++ct) {
#pragma unroll
        for (int r = 0; r < 8; ++r) sc[ct][r] = 0.f;
#pragma unroll
        for (int ka = 0; ka < 4; ++ka) {
          v16bf kb = bfrag_bf(Kc + ct * 16 * 128, 128, ka * 32);
          sc[ct] = wmma_bf16(ya[ka], kb, sc[ct]);
        }
      }
      float actc[4];
#pragma unroll
      for (int ct = 0; ct < 4; ++ct) actc[ct] = shS[ct * 16 + lr] > 0.f ? 1.f : 0.f;
#pragma unroll
      for (int ct = 0; ct < 4; ++ct)
#pragma unroll
        for (int r = 0; r < 8; ++r) {
          float v = sc[ct][r] * invtau;
          sc[ct][r] = (actc[ct] > 0.f) ? v : NEGV;
        }
#pragma unroll
      for (int r = 0; r < 8; ++r) {
        float cm = fmaxf(fmaxf(sc[0][r], sc[1][r]), fmaxf(sc[2][r], sc[3][r]));
        cm = half_max(cm);
        float newm = fmaxf(rowmax[r], cm);
        float resc = __expf(rowmax[r] - newm);
        rowmax[r] = newm;
        float ps = 0.f;
#pragma unroll
        for (int ct = 0; ct < 4; ++ct) {
          float p = __expf(sc[ct][r] - newm);
          sc[ct][r] = p; ps += p;
        }
        ps = half_sum(ps);
        rowsum[r] = rowsum[r] * resc + ps;
#pragma unroll
        for (int f = 0; f < 8; ++f) acc[f][r] *= resc;
      }
#pragma unroll
      for (int ct = 0; ct < 4; ++ct)
#pragma unroll
        for (int r = 0; r < 8; ++r)
          Pw[(r + 8 * hf) * 64 + ct * 16 + lr] = (__bf16)sc[ct][r];
      v16bf pa[2];
#pragma unroll
      for (int kc = 0; kc < 2; ++kc) pa[kc] = afrag_bf(Pw, 64, kc * 32);
#pragma unroll
      for (int dt = 0; dt < 8; ++dt)
#pragma unroll
        for (int kc = 0; kc < 2; ++kc) {
          v16bf vb = bfrag_bf(Vt + dt * 16 * 64, 64, kc * 32);
          acc[dt] = wmma_bf16(pa[kc], vb, acc[dt]);
        }
    }
    float rsin[8];
#pragma unroll
    for (int r = 0; r < 8; ++r) rsin[r] = hact / fmaxf(rowsum[r], 1e-30f);
#pragma unroll
    for (int f = 0; f < 8; ++f)
#pragma unroll
      for (int r = 0; r < 8; ++r) {
        float dl = acc[f][r] * rsin[r];
        float g = 1.f / (1.f + __expf(-(w1v[f] * yc[f][r] + w2v[f] * dl + gbv[f])));
        yc[f][r] += g * dl;
      }
  }
#pragma unroll
  for (int f = 0; f < 8; ++f)
#pragma unroll
    for (int r = 0; r < 8; ++r) {
      int n = row0 + r + 8 * hf, d = f * 16 + lr;
      y_em[((size_t)(bs * N_ + n) * B_ + b) * D_ + d] =
          yc[f][r] - seedB[(size_t)n * D_ + d];
    }
}

// -------------------- novelty / recon + route-softmax stats ---------------
__global__ __launch_bounds__(256, 1)
void k_nov(const float* __restrict__ w_cand, const __bf16* __restrict__ wnb,
           const __bf16* __restrict__ emKb, const __bf16* __restrict__ emVt,
           const float* __restrict__ emS, const float* __restrict__ raw_tau,
           const float* __restrict__ raw_tau_w, const float* __restrict__ hact_ws,
           const float* __restrict__ smag, float* __restrict__ nov_ws,
           float* __restrict__ rmw_ws, float* __restrict__ rsw_ws) {
  __shared__ __align__(16) __bf16 sh[24576];
  __shared__ float shS[64];
  const int blk = blockIdx.x;
  const int nt = blk & 7, b = (blk >> 3) & 7, bs = blk >> 6;
  const int wave = (int)(threadIdx.x >> 5);
  const int lr = lane_id() & 15, hf = lane_id() >> 4;
  const int row0 = nt * 128 + wave * 16;
  const float invtau = 1.0f / softplus01(raw_tau[b]);
  const float invtw = 1.0f / softplus01(raw_tau_w[b]);
  const float hact = hact_ws[bs * B_ + b];
  const __bf16* Kg = emKb + (size_t)(bs * B_ + b) * M_ * D_;
  const __bf16* Vg = emVt + (size_t)(bs * B_ + b) * D_ * M_;
  const float* Sb = emS + (size_t)(bs * B_ + b) * M_;
  __bf16* Kc = sh;
  __bf16* Vt = sh + 8192;
  __bf16* Pw = sh + 16384 + wave * 1024;

  v16bf aw[4];
#pragma unroll
  for (int ka = 0; ka < 4; ++ka)
    aw[ka] = afrag_bf(wnb + (size_t)(bs * N_ + row0) * D_, D_, ka * 32);

  float rowmax[8], rowsum[8], rmw[8], rsw[8];
  v8f acc[8];
#pragma unroll
  for (int r = 0; r < 8; ++r) { rowmax[r] = -INFINITY; rowsum[r] = 0.f; rmw[r] = -INFINITY; rsw[r] = 0.f; }
#pragma unroll
  for (int f = 0; f < 8; ++f)
#pragma unroll
    for (int r = 0; r < 8; ++r) acc[f][r] = 0.f;

  for (int mc = 0; mc < M_; mc += 64) {
    __syncthreads();
    for (int u = (int)threadIdx.x; u < 1024; u += 256) {
      cp16_g2l(Kc + u * 8, Kg + (size_t)mc * D_ + u * 8);
      int d = u >> 3, q = u & 7;
      cp16_g2l(Vt + d * 64 + q * 8, Vg + (size_t)d * M_ + mc + q * 8);
    }
    if (threadIdx.x < 64) shS[threadIdx.x] = Sb[mc + threadIdx.x];
    stage_fence();

    v8f sc[4];
#pragma unroll
    for (int ct = 0; ct < 4; ++ct) {
#pragma unroll
      for (int r = 0; r < 8; ++r) sc[ct][r] = 0.f;
#pragma unroll
      for (int ka = 0; ka < 4; ++ka) {
        v16bf kb = bfrag_bf(Kc + ct * 16 * 128, 128, ka * 32);
        sc[ct] = wmma_bf16(aw[ka], kb, sc[ct]);
      }
    }
    float actc[4];
#pragma unroll
    for (int ct = 0; ct < 4; ++ct) actc[ct] = shS[ct * 16 + lr] > 0.f ? 1.f : 0.f;

    // route (tau_w) online stats
#pragma unroll
    for (int r = 0; r < 8; ++r) {
      float vw[4];
#pragma unroll
      for (int ct = 0; ct < 4; ++ct)
        vw[ct] = (actc[ct] > 0.f) ? sc[ct][r] * invtw : NEGV;
      float cm = fmaxf(fmaxf(vw[0], vw[1]), fmaxf(vw[2], vw[3]));
      cm = half_max(cm);
      float newm = fmaxf(rmw[r], cm);
      float resc = __expf(rmw[r] - newm);
      float ps = 0.f;
#pragma unroll
      for (int ct = 0; ct < 4; ++ct) ps += __expf(vw[ct] - newm);
      ps = half_sum(ps);
      rsw[r] = rsw[r] * resc + ps;
      rmw[r] = newm;
    }
    // recon attention (tau), flash accumulate
#pragma unroll
    for (int ct = 0; ct < 4; ++ct)
#pragma unroll
      for (int r = 0; r < 8; ++r) {
        float v = sc[ct][r] * invtau;
        sc[ct][r] = (actc[ct] > 0.f) ? v : NEGV;
      }
#pragma unroll
    for (int r = 0; r < 8; ++r) {
      float cm = fmaxf(fmaxf(sc[0][r], sc[1][r]), fmaxf(sc[2][r], sc[3][r]));
      cm = half_max(cm);
      float newm = fmaxf(rowmax[r], cm);
      float resc = __expf(rowmax[r] - newm);
      rowmax[r] = newm;
      float ps = 0.f;
#pragma unroll
      for (int ct = 0; ct < 4; ++ct) {
        float p = __expf(sc[ct][r] - newm);
        sc[ct][r] = p; ps += p;
      }
      ps = half_sum(ps);
      rowsum[r] = rowsum[r] * resc + ps;
#pragma unroll
      for (int f = 0; f < 8; ++f) acc[f][r] *= resc;
    }
#pragma unroll
    for (int ct = 0; ct < 4; ++ct)
#pragma unroll
      for (int r = 0; r < 8; ++r)
        Pw[(r + 8 * hf) * 64 + ct * 16 + lr] = (__bf16)sc[ct][r];
    v16bf pa[2];
#pragma unroll
    for (int kc = 0; kc < 2; ++kc) pa[kc] = afrag_bf(Pw, 64, kc * 32);
#pragma unroll
    for (int dt = 0; dt < 8; ++dt)
#pragma unroll
      for (int kc = 0; kc < 2; ++kc) {
        v16bf vb = bfrag_bf(Vt + dt * 16 * 64, 64, kc * 32);
        acc[dt] = wmma_bf16(pa[kc], vb, acc[dt]);
      }
  }
  float rsin[8];
#pragma unroll
  for (int r = 0; r < 8; ++r) rsin[r] = hact / fmaxf(rowsum[r], 1e-30f);
  float e2[8];
#pragma unroll
  for (int r = 0; r < 8; ++r) {
    float s = 0.f;
    int n = row0 + r + 8 * hf;
#pragma unroll
    for (int f = 0; f < 8; ++f) {
      float rec = acc[f][r] * rsin[r];
      float df = w_cand[((size_t)bs * N_ + n) * D_ + f * 16 + lr] - rec;
      s += df * df;
    }
    e2[r] = half_sum(s);
  }
  if (lr == 0) {
#pragma unroll
    for (int r = 0; r < 8; ++r) {
      int n = row0 + r + 8 * hf;
      int id = (bs * B_ + b) * N_ + n;
      nov_ws[id] = 0.5f * smag[bs * N_ + n] + 0.5f * sqrtf(e2[r]);
      rmw_ws[id] = rmw[r];
      rsw_ws[id] = rsw[r];
    }
  }
}

// -------------------- delta_em outer product --------------------
__global__ void k_delta_em(const float* __restrict__ w_cand,
                           const float* __restrict__ nov, float* __restrict__ out) {
  size_t idx = (size_t)blockIdx.x * 256 + threadIdx.x;
  int d = (int)(idx & 127);
  size_t t = idx >> 7;
  int b = (int)(t & 7);
  size_t t2 = t >> 3;
  int n = (int)(t2 & 1023);
  int bs = (int)(t2 >> 10);
  out[idx] = nov[(size_t)(bs * B_ + b) * N_ + n] *
             w_cand[((size_t)bs * N_ + n) * D_ + d];
}

// -------------------- routed memory update (WMMA over N) ------------------
__global__ __launch_bounds__(256, 1)
void k_update(const __bf16* __restrict__ wnb, const __bf16* __restrict__ wnt,
              const __bf16* __restrict__ wct, const __bf16* __restrict__ emKb,
              const float* __restrict__ emK, const float* __restrict__ emV,
              const float* __restrict__ emS, const float* __restrict__ emAge,
              const float* __restrict__ gem, const float* __restrict__ raw_tau_w,
              const float* __restrict__ nov_ws, const float* __restrict__ rmw_ws,
              const float* __restrict__ rsw_ws, float* __restrict__ outK,
              float* __restrict__ outV, float* __restrict__ outS,
              float* __restrict__ outAge) {
  __shared__ __align__(16) __bf16 sh[24576];  // Wt | Wc | Pw  (48KB)
  __shared__ float shn[64], shm[64], shs[64];
  const int blk = blockIdx.x;
  const int mt = blk & 3, b = (blk >> 2) & 7, bs = blk >> 5;
  const int wave = (int)(threadIdx.x >> 5);
  const int lr = lane_id() & 15, hf = lane_id() >> 4;
  const int mrow0 = mt * 128 + wave * 16;
  const float invtw = 1.0f / softplus01(raw_tau_w[b]);
  const float gv = gem[bs * B_ + b];
  const float* Sb = emS + (size_t)(bs * B_ + b) * M_;
  const float* Ab = emAge + (size_t)(bs * B_ + b) * M_;
  const __bf16* Wg = wnt + (size_t)bs * D_ * N_;  // [d][n]
  const __bf16* Cg = wct + (size_t)bs * D_ * N_;  // [d][n]
  __bf16* Wt = sh;
  __bf16* Wc = sh + 8192;
  __bf16* Pw = sh + 16384 + wave * 1024;

  v16bf aK[4];
#pragma unroll
  for (int ka = 0; ka < 4; ++ka)
    aK[ka] = afrag_bf(emKb + ((size_t)(bs * B_ + b) * M_ + mrow0) * D_, D_, ka * 32);
  float srow[8];
#pragma unroll
  for (int r = 0; r < 8; ++r) srow[r] = Sb[mrow0 + r + 8 * hf];

  v8f accK[8], accV[8];
  float part[8];
#pragma unroll
  for (int r = 0; r < 8; ++r) part[r] = 0.f;
#pragma unroll
  for (int f = 0; f < 8; ++f)
#pragma unroll
    for (int r = 0; r < 8; ++r) { accK[f][r] = 0.f; accV[f][r] = 0.f; }

  for (int n0 = 0; n0 < N_; n0 += 64) {
    __syncthreads();
    for (int u = (int)threadIdx.x; u < 1024; u += 256) {
      int d = u >> 3, q = u & 7;
      cp16_g2l(Wt + d * 64 + q * 8, Wg + (size_t)d * N_ + n0 + q * 8);
      cp16_g2l(Wc + d * 64 + q * 8, Cg + (size_t)d * N_ + n0 + q * 8);
    }
    if (threadIdx.x < 64) {
      int id = (bs * B_ + b) * N_ + n0 + (int)threadIdx.x;
      shn[threadIdx.x] = nov_ws[id];
      shm[threadIdx.x] = rmw_ws[id];
      shs[threadIdx.x] = rsw_ws[id];
    }
    stage_fence();
#pragma unroll
    for (int ct = 0; ct < 4; ++ct) {
      v8f s;
#pragma unroll
      for (int r = 0; r < 8; ++r) s[r] = 0.f;
#pragma unroll
      for (int ka = 0; ka < 4; ++ka) {
        v16bf bw = bfrag_bf(wnb + (size_t)(bs * N_ + n0 + ct * 16) * D_, D_, ka * 32);
        s = wmma_bf16(aK[ka], bw, s);
      }
      float nvv = shn[ct * 16 + lr];
      float rmv = shm[ct * 16 + lr];
      float rsv = fmaxf(shs[ct * 16 + lr], 1e-30f);
#pragma unroll
      for (int r = 0; r < 8; ++r) {
        float v = (srow[r] > 0.f) ? s[r] * invtw : NEGV;
        float p = __expf(v - rmv) / rsv * nvv;
        part[r] += p;
        Pw[(r + 8 * hf) * 64 + ct * 16 + lr] = (__bf16)p;
      }
    }
    v16bf pa[2];
#pragma unroll
    for (int kc = 0; kc < 2; ++kc) pa[kc] = afrag_bf(Pw, 64, kc * 32);
#pragma unroll
    for (int dt = 0; dt < 8; ++dt)
#pragma unroll
      for (int kc = 0; kc < 2; ++kc) {
        accK[dt] = wmma_bf16(pa[kc], bfrag_bf(Wt + dt * 16 * 64, 64, kc * 32), accK[dt]);
        accV[dt] = wmma_bf16(pa[kc], bfrag_bf(Wc + dt * 16 * 64, 64, kc * 32), accV[dt]);
      }
  }
  float alpha[8], dinv[8];
#pragma unroll
  for (int r = 0; r < 8; ++r) {
    float t = half_sum(part[r]);
    alpha[r] = fminf(gv * (t / (float)N_), 1.f);
    dinv[r] = 1.f / fmaxf(t, 1e-8f);
  }
  float kin[8];
#pragma unroll
  for (int r = 0; r < 8; ++r) {
    float s = 0.f;
#pragma unroll
    for (int dt = 0; dt < 8; ++dt) { float u = accK[dt][r] * dinv[r]; s += u * u; }
    kin[r] = 1.f / fmaxf(sqrtf(half_sum(s)), 1e-12f);
  }
#pragma unroll
  for (int dt = 0; dt < 8; ++dt)
#pragma unroll
    for (int r = 0; r < 8; ++r) {
      int m = mrow0 + r + 8 * hf, d = dt * 16 + lr;
      size_t gi = (size_t)(bs * B_ + b) * M_ * D_ + (size_t)m * D_ + d;
      float a = alpha[r];
      outK[gi] = (1.f - a) * emK[gi] + a * (accK[dt][r] * dinv[r] * kin[r]);
      outV[gi] = (1.f - a) * emV[gi] + a * (accV[dt][r] * dinv[r]);
    }
  if (lr == 0) {
#pragma unroll
    for (int r = 0; r < 8; ++r) {
      int m = mrow0 + r + 8 * hf;
      size_t gi = (size_t)(bs * B_ + b) * M_ + m;
      float a = alpha[r];
      outS[gi] = fminf(fmaxf(srow[r] + a, 0.f), 3.0f);
      outAge[gi] = Ab[m] * (1.f - a);
    }
  }
}

// -------------------- S budget rescale --------------------
__global__ void k_budget(float* __restrict__ outS) {
  __shared__ float red[8];
  __shared__ float scl;
  float* p = outS + (size_t)blockIdx.x * M_;
  float s = 0.f;
  for (int i = (int)threadIdx.x; i < M_; i += 256) s += p[i];
#pragma unroll
  for (int m = 1; m < 32; m <<= 1) s += __shfl_xor(s, m, 32);
  if ((threadIdx.x & 31) == 0) red[threadIdx.x >> 5] = s;
  __syncthreads();
  if (threadIdx.x == 0) {
    float t = 0.f;
    for (int i = 0; i < 8; ++i) t += red[i];
    scl = fminf(1.f, 32.0f / fmaxf(t, 1e-8f));
  }
  __syncthreads();
  float sc = scl;
  for (int i = (int)threadIdx.x; i < M_; i += 256) p[i] *= sc;
}

extern "C" void kernel_launch(void* const* d_in, const int* in_sizes, int n_in,
                              void* d_out, int out_size, void* d_ws, size_t ws_size,
                              hipStream_t stream) {
  (void)in_sizes; (void)n_in; (void)out_size; (void)ws_size;
  const float* seed      = (const float*)d_in[0];
  const float* w_cand    = (const float*)d_in[1];
  const float* surprise  = (const float*)d_in[2];
  const float* g_em      = (const float*)d_in[3];
  const float* em_K      = (const float*)d_in[4];
  const float* em_V      = (const float*)d_in[5];
  const float* em_S      = (const float*)d_in[6];
  const float* em_age    = (const float*)d_in[7];
  const float* w1        = (const float*)d_in[8];
  const float* w2        = (const float*)d_in[9];
  const float* gate_bias = (const float*)d_in[10];
  const float* raw_tau   = (const float*)d_in[11];
  const float* raw_tau_w = (const float*)d_in[12];
  float* out = (float*)d_out;
  float* ws  = (float*)d_ws;

  // f32 workspace
  float* wn   = ws;                 // BS*N*D     = 1048576
  float* smag = wn + 1048576;       // BS*N       = 8192
  float* hact = smag + 8192;        // BS*B       = 64
  float* nov  = hact + 64;          // BS*B*N     = 65536
  float* rmw  = nov + 65536;
  float* rsw  = rmw + 65536;
  // bf16 workspace (16B-aligned: 1253440 floats from base)
  __bf16* bws  = (__bf16*)(ws + 1253440);
  __bf16* emKb = bws;               // BS*B*M*D row-major
  __bf16* emVt = emKb + 4194304;    // BS*B*[D][M]
  __bf16* wnb  = emVt + 4194304;    // BS*[N][D]
  __bf16* wnt  = wnb + 1048576;     // BS*[D][N]
  __bf16* wct  = wnt + 1048576;     // BS*[D][N]

  float* y_em     = out;
  float* delta_em = out + 8388608;
  float* nK       = out + 16777216;
  float* nV       = out + 20971520;
  float* nS       = out + 25165824;
  float* nAge     = out + 25198592;

  k_prep<<<BS_ * N_, 128, 0, stream>>>(w_cand, surprise, wn, smag);
  k_hasact<<<BS_ * B_, 256, 0, stream>>>(em_S, hact);
  k_cvt_copy<<<16384, 256, 0, stream>>>(em_K, emKb, BS_ * B_ * M_ * D_);
  k_cvt_tr<<<dim3(256, BS_ * B_), 256, 0, stream>>>(em_V, emVt, M_, D_);
  k_cvt_copy<<<4096, 256, 0, stream>>>(wn, wnb, BS_ * N_ * D_);
  k_cvt_tr<<<dim3(512, BS_), 256, 0, stream>>>(wn, wnt, N_, D_);
  k_cvt_tr<<<dim3(512, BS_), 256, 0, stream>>>(w_cand, wct, N_, D_);

  k_attn<<<BS_ * B_ * (N_ / 128), 256, 0, stream>>>(seed, emKb, emVt, em_S, w1, w2,
                                                    gate_bias, raw_tau, hact, y_em);
  k_nov<<<BS_ * B_ * (N_ / 128), 256, 0, stream>>>(w_cand, wnb, emKb, emVt, em_S,
                                                   raw_tau, raw_tau_w, hact, smag,
                                                   nov, rmw, rsw);
  k_delta_em<<<(BS_ * N_ * B_ * D_) / 256, 256, 0, stream>>>(w_cand, nov, delta_em);
  k_update<<<BS_ * B_ * (M_ / 128), 256, 0, stream>>>(wnb, wnt, wct, emKb, em_K,
                                                      em_V, em_S, em_age, g_em,
                                                      raw_tau_w, nov, rmw, rsw,
                                                      nK, nV, nS, nAge);
  k_budget<<<BS_ * B_, 256, 0, stream>>>(nS);
}
